// GNN_VirtualNode_19069654794767
// MI455X (gfx1250) — compile-verified, matrix operand order checked by reading, not run on hardware
//
#include <hip/hip_runtime.h>
#include <hip/hip_bf16.h>

#define NN 100000
#define NE 300000
#define NG 2000
#define DD 256
#define NL 5
#define NH_ELEMS ((size_t)NN * DD)

typedef __attribute__((ext_vector_type(16))) _Float16 v16h;
typedef __attribute__((ext_vector_type(8)))  _Float16 v8h;
typedef __attribute__((ext_vector_type(4)))  _Float16 v4h;
typedef __attribute__((ext_vector_type(8)))  float    v8f;

union V16 { v16h v; v8h h[2]; };

#if __has_builtin(__builtin_amdgcn_global_load_async_to_lds_b128)
#define HAVE_ASYNC_LDS 1
// Builtin expects: (v4i addrspace(1)*, v4i addrspace(3)*, imm i32, imm i32)
typedef int v4i __attribute__((vector_size(16)));
typedef __attribute__((address_space(1))) v4i* async_gptr;
typedef __attribute__((address_space(3))) v4i* async_lptr;
#else
#define HAVE_ASYNC_LDS 0
#endif

// ---------------------------------------------------------------------------
// Node encoder: h = x @ node_w + node_b   ([NN,20] @ [20,256])
// ---------------------------------------------------------------------------
__global__ __launch_bounds__(256) void k_encode(
    const float* __restrict__ x, const float* __restrict__ w,
    const float* __restrict__ b, float* __restrict__ h) {
  const int node = blockIdx.x;
  const int c = threadIdx.x;               // 256 threads
  float acc = b[c];
  const float* xr = x + node * 20;
  #pragma unroll
  for (int k = 0; k < 20; ++k) acc = fmaf(xr[k], w[k * DD + c], acc);
  h[(size_t)node * DD + c] = acc;
}

// vn[g][d] = vn_w[d]
__global__ __launch_bounds__(256) void k_vn_init(
    const float* __restrict__ vnw, float* __restrict__ vn) {
  vn[(size_t)blockIdx.x * DD + threadIdx.x] = vnw[threadIdx.x];
}

// h[node] += vn[batch[node]]   (float4 per thread)
__global__ __launch_bounds__(256) void k_add_vn(
    float* __restrict__ h, const float* __restrict__ vn,
    const int* __restrict__ batch) {
  const int idx = blockIdx.x * blockDim.x + threadIdx.x;   // NN*64 threads
  const int node = idx >> 6;
  if (node >= NN) return;
  const int d4 = (idx & 63) << 2;
  const int g = batch[node];
  float4 hv = *(float4*)(h + (size_t)node * DD + d4);
  const float4 vv = *(const float4*)(vn + (size_t)g * DD + d4);
  hv.x += vv.x; hv.y += vv.y; hv.z += vv.z; hv.w += vv.w;
  *(float4*)(h + (size_t)node * DD + d4) = hv;
}

// agg[dst[e]] += h[src[e]]
__global__ __launch_bounds__(256) void k_scatter_add(
    const float* __restrict__ h, const int* __restrict__ src,
    const int* __restrict__ dst, float* __restrict__ agg) {
  const int idx = blockIdx.x * blockDim.x + threadIdx.x;   // NE*64 threads
  const int e = idx >> 6;
  if (e >= NE) return;
  const int d4 = (idx & 63) << 2;
  const int s = src[e], t = dst[e];
  const float4 v = *(const float4*)(h + (size_t)s * DD + d4);
  float* a = agg + (size_t)t * DD + d4;
  atomicAdd(a + 0, v.x); atomicAdd(a + 1, v.y);
  atomicAdd(a + 2, v.z); atomicAdd(a + 3, v.w);
}

// z = f16(h + agg)
__global__ __launch_bounds__(256) void k_combine_f16(
    const float* __restrict__ h, const float* __restrict__ agg,
    _Float16* __restrict__ z) {
  const int idx = blockIdx.x * blockDim.x + threadIdx.x;   // NH/4 threads
  if ((size_t)idx >= NH_ELEMS / 4) return;
  const float4 a = ((const float4*)h)[idx];
  const float4 b = ((const float4*)agg)[idx];
  v4h o;
  o.x = (_Float16)(a.x + b.x); o.y = (_Float16)(a.y + b.y);
  o.z = (_Float16)(a.z + b.z); o.w = (_Float16)(a.w + b.w);
  ((v4h*)z)[idx] = o;
}

// pool[batch[node]] += h[node]
__global__ __launch_bounds__(256) void k_pool(
    const float* __restrict__ h, const int* __restrict__ batch,
    float* __restrict__ pool) {
  const int idx = blockIdx.x * blockDim.x + threadIdx.x;   // NN*64 threads
  const int node = idx >> 6;
  if (node >= NN) return;
  const int d4 = (idx & 63) << 2;
  const int g = batch[node];
  const float4 v = *(const float4*)(h + (size_t)node * DD + d4);
  float* p = pool + (size_t)g * DD + d4;
  atomicAdd(p + 0, v.x); atomicAdd(p + 1, v.y);
  atomicAdd(p + 2, v.z); atomicAdd(p + 3, v.w);
}

// t = f16(pool + vn)
__global__ __launch_bounds__(256) void k_pool_vn_f16(
    const float* __restrict__ pool, const float* __restrict__ vn,
    _Float16* __restrict__ t) {
  const int idx = blockIdx.x * blockDim.x + threadIdx.x;   // NG*DD threads
  t[idx] = (_Float16)(pool[idx] + vn[idx]);
}

// wt[n][k] = f16(w[k][n])  (B stored column-major => B-fragment load == A-fragment load)
__global__ __launch_bounds__(256) void k_wt(
    const float* __restrict__ w, _Float16* __restrict__ wt) {
  const int idx = blockIdx.x * blockDim.x + threadIdx.x;   // DD*DD threads
  const int n = idx & (DD - 1);
  const int k = idx >> 8;
  wt[(size_t)n * DD + k] = (_Float16)w[(size_t)k * DD + n];
}

// small dense head layer: out[g][c] = act(in[g] . w[:,c] + b[c])
__global__ __launch_bounds__(256) void k_head(
    const float* __restrict__ in, const float* __restrict__ w,
    const float* __restrict__ b, float* __restrict__ out,
    int K, int N, int relu) {
  const int g = blockIdx.x;
  const int c = threadIdx.x;
  if (c >= N) return;
  float acc = b[c];
  const float* ir = in + (size_t)g * K;
  for (int k = 0; k < K; ++k) acc = fmaf(ir[k], w[k * N + c], acc);
  if (relu) acc = fmaxf(acc, 0.f);
  out[(size_t)g * N + c] = acc;
}

// ---------------------------------------------------------------------------
// WMMA GEMM:  out = act(A[M,256] @ W[256,256] + bias)  with optional BatchNorm.
// A is f16 row-major; Wt is f16 N-major (W transposed) and staged in LDS
// (async global->LDS when available, else b128 copies through VGPRs).
// Block: 256 threads (8 waves). Wave w -> rows [blk*128 + w*16, +16), all 256
// cols in two 128-col halves. K loop: 8 x v_wmma_f32_16x16x32_f16 per half.
// __launch_bounds__(256, 1): 8 waves/block known at compile time -> full VGPR
// budget, no scratch spills (A fragments are reloaded from global for every
// K-step of each N-half).
// MODE 0: relu           -> f16 out
// MODE 1: bn+leaky(0.1)  -> f32 out
// MODE 2: bn+relu        -> f16 out
// MODE 3: bn+relu        -> f32 out
// ---------------------------------------------------------------------------
template <int MODE>
__global__ __launch_bounds__(256, 1) void k_gemm(
    const _Float16* __restrict__ A, const _Float16* __restrict__ Wt,
    const float* __restrict__ bias,
    const float* __restrict__ bng, const float* __restrict__ bnb,
    const float* __restrict__ bnm, const float* __restrict__ bnv,
    int M, _Float16* __restrict__ outH, float* __restrict__ outF) {
  extern __shared__ _Float16 lwt[];   // 256*256 f16 = 128 KB (dynamic LDS)

  // Stage full Wt into LDS.
#if HAVE_ASYNC_LDS
  {
    // Async DMA path: 16B per lane per op, no VGPR round-trip (ASYNCcnt).
    for (int i = threadIdx.x; i < DD * DD / 8; i += 256) {
      __builtin_amdgcn_global_load_async_to_lds_b128(
          (async_gptr)(void*)(const_cast<_Float16*>(Wt) + (size_t)i * 8),
          (async_lptr)(lwt + (size_t)i * 8),
          0, 0);
    }
#if __has_builtin(__builtin_amdgcn_s_wait_asynccnt)
    __builtin_amdgcn_s_wait_asynccnt(0);
#else
    asm volatile("s_wait_asynccnt 0" ::: "memory");
#endif
  }
#else
  {
    const float4* s = (const float4*)Wt;
    float4* d = (float4*)lwt;
    #pragma unroll 4
    for (int i = threadIdx.x; i < DD * DD / 8; i += 256) d[i] = s[i];
  }
#endif
  __syncthreads();

  const int wave   = threadIdx.x >> 5;
  const int lane   = threadIdx.x & 31;
  const int laneM  = lane & 15;            // A row / B col / D col within tile
  const int koff   = (lane >> 4) << 3;     // K offset 0 or 8 (ISA f16 layout)
  const int rowoff = (lane >> 4) << 3;     // D row offset 0 or 8
  const int mtile  = blockIdx.x * 128 + wave * 16;

  int arow = mtile + laneM;
  if (arow >= M) arow = M - 1;             // clamp loads; stores are guarded
  const _Float16* abase = A + (size_t)arow * DD;

  for (int nh = 0; nh < 2; ++nh) {
    const int nbase = nh * 128;
    v8f acc[8] = {};

    #pragma unroll
    for (int kt = 0; kt < 8; ++kt) {
      const int kb = kt * 32;
      V16 a;
      a.h[0] = *(const v8h*)(abase + kb + koff);
      a.h[1] = *(const v8h*)(abase + kb + koff + 16);
      #pragma unroll
      for (int nt = 0; nt < 8; ++nt) {
        const _Float16* bp = lwt + (size_t)(nbase + nt * 16 + laneM) * DD + kb + koff;
        V16 b;
        b.h[0] = *(const v8h*)(bp);
        b.h[1] = *(const v8h*)(bp + 16);
        acc[nt] = __builtin_amdgcn_wmma_f32_16x16x32_f16(
            /*neg_a=*/false, a.v, /*neg_b=*/false, b.v,
            /*c_mod=*/(short)0, acc[nt], /*reuse_a=*/false, /*reuse_b=*/false);
      }
    }

    // Fused epilogue: bias (+BN) + activation, store f16 or f32.
    #pragma unroll
    for (int nt = 0; nt < 8; ++nt) {
      const int col = nbase + nt * 16 + laneM;
      const float bs = bias[col];
      float sc = 1.f, sh = 0.f;
      if (MODE >= 1) {
        const float gg = bng[col], bb = bnb[col], mm = bnm[col], vv = bnv[col];
        sc = gg * rsqrtf(vv + 1e-5f);
        sh = bb - mm * sc;
      }
      #pragma unroll
      for (int r = 0; r < 8; ++r) {
        const int mr = mtile + rowoff + r;
        if (mr < M) {
          float v = acc[nt][r] + bs;
          if (MODE == 0) {
            v = fmaxf(v, 0.f);
          } else {
            v = v * sc + sh;
            if (MODE == 1) v = (v > 0.f) ? v : 0.1f * v;
            else           v = fmaxf(v, 0.f);
          }
          if (MODE == 0 || MODE == 2) outH[(size_t)mr * DD + col] = (_Float16)v;
          else                        outF[(size_t)mr * DD + col] = v;
        }
      }
    }
  }
}

// ---------------------------------------------------------------------------
extern "C" void kernel_launch(void* const* d_in, const int* in_sizes, int n_in,
                              void* d_out, int out_size, void* d_ws, size_t ws_size,
                              hipStream_t stream) {
  const float* x      = (const float*)d_in[0];
  const int*   edge   = (const int*)  d_in[1];
  const int*   src    = edge;
  const int*   dst    = edge + NE;
  const int*   batch  = (const int*)  d_in[2];
  const float* node_w = (const float*)d_in[3];
  const float* node_b = (const float*)d_in[4];
  const float* vn_w   = (const float*)d_in[5];
  const float* gin_w1 = (const float*)d_in[6];
  const float* gin_b1 = (const float*)d_in[7];
  const float* gin_w2 = (const float*)d_in[8];
  const float* gin_b2 = (const float*)d_in[9];
  const float* bn_g   = (const float*)d_in[10];
  const float* bn_b   = (const float*)d_in[11];
  const float* bn_m   = (const float*)d_in[12];
  const float* bn_v   = (const float*)d_in[13];
  const float* vn_w1  = (const float*)d_in[14];
  const float* vn_b1  = (const float*)d_in[15];
  const float* vbn1g  = (const float*)d_in[16];
  const float* vbn1b  = (const float*)d_in[17];
  const float* vbn1m  = (const float*)d_in[18];
  const float* vbn1v  = (const float*)d_in[19];
  const float* vn_w2  = (const float*)d_in[20];
  const float* vn_b2  = (const float*)d_in[21];
  const float* vbn2g  = (const float*)d_in[22];
  const float* vbn2b  = (const float*)d_in[23];
  const float* vbn2m  = (const float*)d_in[24];
  const float* vbn2v  = (const float*)d_in[25];
  const float* hw1    = (const float*)d_in[26];
  const float* hb1    = (const float*)d_in[27];
  const float* hw2    = (const float*)d_in[28];
  const float* hb2    = (const float*)d_in[29];
  const float* hw3    = (const float*)d_in[30];
  const float* hb3    = (const float*)d_in[31];
  float* out = (float*)d_out;

  // Workspace carve-up
  char* p = (char*)d_ws;
  float*    h    = (float*)p;     p += NH_ELEMS * 4;
  float*    agg  = (float*)p;     p += NH_ELEMS * 4;
  _Float16* z0   = (_Float16*)p;  p += NH_ELEMS * 2;
  _Float16* z1   = (_Float16*)p;  p += NH_ELEMS * 2;
  float*    vn   = (float*)p;     p += (size_t)NG * DD * 4;
  float*    pool = (float*)p;     p += (size_t)NG * DD * 4;
  _Float16* t0   = (_Float16*)p;  p += (size_t)NG * DD * 2;
  _Float16* t1   = (_Float16*)p;  p += (size_t)NG * DD * 2;
  _Float16* wtA  = (_Float16*)p;  p += (size_t)DD * DD * 2;
  _Float16* wtB  = (_Float16*)p;  p += (size_t)DD * DD * 2;
  float*    g1   = (float*)p;     p += (size_t)NG * 128 * 4;
  float*    g2   = (float*)p;     p += (size_t)NG * 64 * 4;

  const int B = 256;
  const int GB_NODE = (NN + 127) / 128;                // GEMM grid over nodes
  const int GB_GRPH = (NG + 127) / 128;                // GEMM grid over graphs
  const size_t LDSB = (size_t)DD * DD * sizeof(_Float16);
  const int EW_NODE = (NN * 64 + B - 1) / B;           // elementwise grids
  const int EW_EDGE = (NE * 64 + B - 1) / B;
  const int EW_NH4  = (int)((NH_ELEMS / 4 + B - 1) / B);
  const int EW_GD   = (NG * DD + B - 1) / B;
  const int WT_G    = (DD * DD + B - 1) / B;

  // 1. Node encoding + virtual-node init
  k_encode<<<NN, 256, 0, stream>>>(x, node_w, node_b, h);
  k_vn_init<<<NG, 256, 0, stream>>>(vn_w, vn);

  for (int l = 0; l < NL; ++l) {
    // h += vn[batch]
    k_add_vn<<<EW_NODE, B, 0, stream>>>(h, vn, batch);

    // agg = segment_sum(h[src], dst)
    (void)hipMemsetAsync(agg, 0, NH_ELEMS * 4, stream);
    k_scatter_add<<<EW_EDGE, B, 0, stream>>>(h, src, dst, agg);

    // z0 = f16(h + agg)
    k_combine_f16<<<EW_NH4, B, 0, stream>>>(h, agg, z0);

    // GIN MLP layer 1: z1 = relu(z0 @ W1 + b1)   (f16 out)
    k_wt<<<WT_G, B, 0, stream>>>(gin_w1 + (size_t)l * DD * DD, wtA);
    k_gemm<0><<<GB_NODE, B, LDSB, stream>>>(z0, wtA, gin_b1 + l * DD,
                                            nullptr, nullptr, nullptr, nullptr,
                                            NN, z1, nullptr);

    // GIN MLP layer 2 + BN + leaky_relu: h = leaky(bn(z1 @ W2 + b2))  (f32 out)
    k_wt<<<WT_G, B, 0, stream>>>(gin_w2 + (size_t)l * DD * DD, wtB);
    k_gemm<1><<<GB_NODE, B, LDSB, stream>>>(z1, wtB, gin_b2 + l * DD,
                                            bn_g + l * DD, bn_b + l * DD,
                                            bn_m + l * DD, bn_v + l * DD,
                                            NN, nullptr, h);

    if (l < NL - 1) {
      // t0 = f16(segment_sum(h, batch) + vn)
      (void)hipMemsetAsync(pool, 0, (size_t)NG * DD * 4, stream);
      k_pool<<<EW_NODE, B, 0, stream>>>(h, batch, pool);
      k_pool_vn_f16<<<EW_GD, B, 0, stream>>>(pool, vn, t0);

      // VN MLP1: t1 = relu(bn1(t0 @ vn_w1 + vn_b1))  (f16 out)
      k_wt<<<WT_G, B, 0, stream>>>(vn_w1 + (size_t)l * DD * DD, wtA);
      k_gemm<2><<<GB_GRPH, B, LDSB, stream>>>(t0, wtA, vn_b1 + l * DD,
                                              vbn1g + l * DD, vbn1b + l * DD,
                                              vbn1m + l * DD, vbn1v + l * DD,
                                              NG, t1, nullptr);

      // VN MLP2: vn = relu(bn2(t1 @ vn_w2 + vn_b2))  (f32 out)
      k_wt<<<WT_G, B, 0, stream>>>(vn_w2 + (size_t)l * DD * DD, wtB);
      k_gemm<3><<<GB_GRPH, B, LDSB, stream>>>(t1, wtB, vn_b2 + l * DD,
                                              vbn2g + l * DD, vbn2b + l * DD,
                                              vbn2m + l * DD, vbn2v + l * DD,
                                              NG, nullptr, vn);
    }
  }

  // Global add pool
  (void)hipMemsetAsync(pool, 0, (size_t)NG * DD * 4, stream);
  k_pool<<<EW_NODE, B, 0, stream>>>(h, batch, pool);

  // Head MLP
  k_head<<<NG, 128, 0, stream>>>(pool, hw1, hb1, g1, 256, 128, 1);
  k_head<<<NG, 64,  0, stream>>>(g1,   hw2, hb2, g2, 128, 64, 1);
  k_head<<<NG, 32,  0, stream>>>(g2,   hw3, hb3, out, 64, 11, 0);
}
